// PointNetSetAbstractionMsg_39316130627753
// MI455X (gfx1250) — matrix operand order repaired and checked by reading, not compiled, not run on hardware
//
#include <hip/hip_runtime.h>
#include <hip/hip_bf16.h>

// ---------------------------------------------------------------------------
// PointNet++ Set Abstraction (MSG) for gfx1250 (MI455X).
// Pipeline: FPS -> 64-NN (LDS-resident distance rows, never materialize the
// [B,S,N] matrix) -> group (f16) -> per-layer [ WMMA f16 GEMM + f32-atomic
// batch stats -> BN(train)+ReLU in place ] -> max over K.
// Matrix ops use v_wmma_f32_16x16x32_f16 with the documented CDNA5 VGPR
// layouts (A interleaved K 0-7/16-23 per lane-half; B same interleaved K chunks
// per lane group; D: m = r + 8*(lane>>4), n = lane&15).
// ---------------------------------------------------------------------------

typedef __attribute__((ext_vector_type(16))) _Float16 v16h;
typedef __attribute__((ext_vector_type(8)))  _Float16 v8h;
typedef __attribute__((ext_vector_type(8)))  float    v8f;

constexpr int B_ = 8;
constexpr int N_ = 16384;
constexpr int S_ = 1024;
constexpr int INCH_ = 32;
constexpr float EPS_ = 1e-5f;
constexpr float BIGF = 3.4e38f;

// ---------------------------------------------------------------------------
// 1) Farthest point sampling: one workgroup per batch, dist[] lives in LDS.
//    Matches reference: dist init +inf, farthest init 0, idx recorded pre-step.
// ---------------------------------------------------------------------------
__global__ void fps_kernel(const float* __restrict__ xyz,
                           int* __restrict__ idx,
                           float* __restrict__ nxw,        // [B,S,3] centroids
                           float* __restrict__ out_xyz) {  // d_out [B,3,S]
    const int b = blockIdx.x;
    const int t = threadIdx.x;
    extern __shared__ char smem[];
    float* dist = (float*)smem;                       // N_ floats
    float* rval = (float*)(smem + (size_t)N_ * 4);    // 1024 floats
    int*   ridx = (int*)(smem + (size_t)N_ * 4 + 4096);
    __shared__ int s_far;

    const float* xp = xyz + (size_t)b * 3 * N_;
    for (int i = t; i < N_; i += blockDim.x) dist[i] = BIGF;
    if (t == 0) s_far = 0;
    __syncthreads();

    for (int s = 0; s < S_; ++s) {
        const int far = s_far;
        const float sx = xp[far], sy = xp[N_ + far], sz = xp[2 * N_ + far];
        if (t == 0) {
            idx[b * S_ + s] = far;
            float* nw = nxw + ((size_t)b * S_ + s) * 3;
            nw[0] = sx; nw[1] = sy; nw[2] = sz;
            out_xyz[((size_t)b * 3 + 0) * S_ + s] = sx;
            out_xyz[((size_t)b * 3 + 1) * S_ + s] = sy;
            out_xyz[((size_t)b * 3 + 2) * S_ + s] = sz;
        }
        float bv = -1.0f; int bi = 0;
        for (int i = t; i < N_; i += blockDim.x) {
            float dx = xp[i] - sx, dy = xp[N_ + i] - sy, dz = xp[2 * N_ + i] - sz;
            float d = dx * dx + dy * dy + dz * dz;
            float nd = fminf(dist[i], d);
            dist[i] = nd;
            if (nd > bv) { bv = nd; bi = i; }
        }
        rval[t] = bv; ridx[t] = bi;
        __syncthreads();
        for (int off = blockDim.x >> 1; off > 0; off >>= 1) {
            if (t < off) {
                float ov = rval[t + off]; int oi = ridx[t + off];
                if (ov > rval[t] || (ov == rval[t] && oi < ridx[t])) { rval[t] = ov; ridx[t] = oi; }
            }
            __syncthreads();
        }
        if (t == 0) s_far = ridx[0];
        __syncthreads();
    }
}

// ---------------------------------------------------------------------------
// 2) 64-NN per centroid (sorted ascending): distance row in LDS + 64 argmin
//    passes with masking. First 32 entries serve the K=32 scale.
// ---------------------------------------------------------------------------
__global__ void knn_kernel(const float* __restrict__ xyz,
                           const float* __restrict__ nxw,
                           int* __restrict__ gidx) {
    const int bs = blockIdx.x;            // b*S + s
    const int b  = bs / S_;
    const int t  = threadIdx.x;
    extern __shared__ char smem[];
    float* dist = (float*)smem;                       // N_ floats
    float* rval = (float*)(smem + (size_t)N_ * 4);    // 256 floats
    int*   ridx = (int*)(smem + (size_t)N_ * 4 + 1024);

    const float* xp = xyz + (size_t)b * 3 * N_;
    const float cx = nxw[(size_t)bs * 3 + 0];
    const float cy = nxw[(size_t)bs * 3 + 1];
    const float cz = nxw[(size_t)bs * 3 + 2];
    for (int i = t; i < N_; i += blockDim.x) {
        float dx = xp[i] - cx, dy = xp[N_ + i] - cy, dz = xp[2 * N_ + i] - cz;
        dist[i] = dx * dx + dy * dy + dz * dz;
    }
    __syncthreads();
    for (int j = 0; j < 64; ++j) {
        float bv = BIGF; int bi = N_;
        for (int i = t; i < N_; i += blockDim.x) {
            float d = dist[i];
            if (d < bv) { bv = d; bi = i; }
        }
        rval[t] = bv; ridx[t] = bi;
        __syncthreads();
        for (int off = blockDim.x >> 1; off > 0; off >>= 1) {
            if (t < off) {
                float ov = rval[t + off]; int oi = ridx[t + off];
                if (ov < rval[t] || (ov == rval[t] && oi < ridx[t])) { rval[t] = ov; ridx[t] = oi; }
            }
            __syncthreads();
        }
        if (t == 0) { gidx[(size_t)bs * 64 + j] = ridx[0]; dist[ridx[0]] = BIGF; }
        __syncthreads();
    }
}

// ---------------------------------------------------------------------------
// 3) Group: build f16 layer-0 activations [M, 64] (35 real ch, zero padded).
// ---------------------------------------------------------------------------
__global__ void group_kernel(const float* __restrict__ xyz,
                             const float* __restrict__ points,
                             const int* __restrict__ gidx,
                             const float* __restrict__ nxw,
                             _Float16* __restrict__ X, int K) {
    size_t tid = (size_t)blockIdx.x * blockDim.x + threadIdx.x;
    size_t total = (size_t)B_ * S_ * K * 64;
    if (tid >= total) return;
    const int c = (int)(tid & 63);
    const size_t m = tid >> 6;
    const int k = (int)(m % K);
    const size_t bs = m / K;
    const int b = (int)(bs / S_);
    const int g = gidx[bs * 64 + k];
    float v;
    if (c < INCH_) {
        v = points[((size_t)b * INCH_ + c) * N_ + g];
    } else if (c < 35) {
        v = xyz[((size_t)b * 3 + (c - 32)) * N_ + g] - nxw[bs * 3 + (c - 32)];
    } else {
        v = 0.0f;
    }
    X[m * 64 + c] = (_Float16)v;
}

// ---------------------------------------------------------------------------
// 4) Weight prep: f32 [O, Cin] -> f16 [O, Cp] (zero-padded K dim).
// ---------------------------------------------------------------------------
__global__ void wprep_kernel(const float* __restrict__ W, _Float16* __restrict__ dst,
                             int O, int Cin, int Cp) {
    int tid = blockIdx.x * blockDim.x + threadIdx.x;
    if (tid >= O * Cp) return;
    int c = tid % Cp, o = tid / Cp;
    dst[tid] = (_Float16)(c < Cin ? W[o * Cin + c] : 0.0f);
}

__global__ void zero_kernel(float* p, int n) {
    int t = blockIdx.x * blockDim.x + threadIdx.x;
    if (t < n) p[t] = 0.0f;
}

// ---------------------------------------------------------------------------
// 5) WMMA GEMM: Y[m,o] = sum_c X[m,c]*W[o,c]; one wave per 16-row M tile,
//    all NOT output tiles accumulated in registers (A loaded once).
//    Also accumulates per-channel sum / sumsq (f32 atomics) for BatchNorm.
// ---------------------------------------------------------------------------
template <int NOT>
__global__ void __launch_bounds__(128)
wmma_gemm_kernel(const _Float16* __restrict__ A, const _Float16* __restrict__ Wf,
                 _Float16* __restrict__ Y, float* __restrict__ stats,
                 int M, int Cp) {
    const int O    = NOT * 16;
    const int lane = threadIdx.x & 31;
    const int wave = threadIdx.x >> 5;
    const int mt   = blockIdx.x * 4 + wave;      // M tile index
    if (mt >= (M >> 4)) return;                  // uniform per wave: EXEC stays full
    const int r16  = lane & 15;
    const int half = lane >> 4;

    // A: lane-half picks interleaved K chunks {h*8 .. h*8+7, h*8+16 .. h*8+23}
    const _Float16* arow = A + (size_t)(mt * 16 + r16) * Cp + half * 8;

    v8f acc[NOT];
#pragma unroll
    for (int ot = 0; ot < NOT; ++ot) acc[ot] = (v8f){};

    for (int kk = 0; kk < Cp; kk += 32) {
        v8h a0 = *(const v8h*)(arow + kk);
        v8h a1 = *(const v8h*)(arow + kk + 16);
        v16h a;
#pragma unroll
        for (int i = 0; i < 8; ++i) { a[i] = a0[i]; a[i + 8] = a1[i]; }
#pragma unroll
        for (int ot = 0; ot < NOT; ++ot) {
            // B: lane holds col n=r16; lane-half picks interleaved K chunks like A
            const _Float16* brow = Wf + (size_t)(ot * 16 + r16) * Cp + half * 8 + kk;
            v8h b0 = *(const v8h*)(brow);
            v8h b1 = *(const v8h*)(brow + 16);
            v16h bm;
#pragma unroll
            for (int i = 0; i < 8; ++i) { bm[i] = b0[i]; bm[i + 8] = b1[i]; }
            acc[ot] = __builtin_amdgcn_wmma_f32_16x16x32_f16(
                false, a, false, bm, (short)0, acc[ot], false, false);
        }
    }

#pragma unroll
    for (int ot = 0; ot < NOT; ++ot) {
        const int o = ot * 16 + r16;
        float s = 0.0f, sq = 0.0f;
#pragma unroll
        for (int r = 0; r < 8; ++r) {
            const int m = mt * 16 + half * 8 + r;    // D layout: m = r + 8*half
            const float v = acc[ot][r];
            Y[(size_t)m * O + o] = (_Float16)v;
            s += v; sq += v * v;
        }
        atomicAdd(&stats[o], s);
        atomicAdd(&stats[128 + o], sq);
    }
}

// ---------------------------------------------------------------------------
// 6) BatchNorm (train-mode batch stats; bias cancels) + ReLU, in place, f16.
// ---------------------------------------------------------------------------
__global__ void bn_relu_kernel(_Float16* __restrict__ Y, const float* __restrict__ stats,
                               const float* __restrict__ gamma, const float* __restrict__ beta,
                               int M, int O) {
    size_t tid = (size_t)blockIdx.x * blockDim.x + threadIdx.x;
    if (tid >= (size_t)M * O) return;
    const int o = (int)(tid % O);
    const float mtot = (float)M;
    const float mean = stats[o] / mtot;
    const float var  = stats[128 + o] / mtot - mean * mean;
    const float inv  = rsqrtf(var + EPS_);
    const float sc   = gamma[o] * inv;
    const float sh   = beta[o] - mean * sc;
    const float v    = sc * (float)Y[tid] + sh;
    Y[tid] = (_Float16)(v > 0.0f ? v : 0.0f);
}

// ---------------------------------------------------------------------------
// 7) Max over K -> d_out new_points [B, 192, S].
// ---------------------------------------------------------------------------
__global__ void maxk_kernel(const _Float16* __restrict__ X, float* __restrict__ out,
                            int K, int O, int chbase) {
    size_t tid = (size_t)blockIdx.x * blockDim.x + threadIdx.x;
    size_t total = (size_t)B_ * S_ * O;
    if (tid >= total) return;
    const int o = (int)(tid % O);
    const size_t bs = tid / O;
    const int b = (int)(bs / S_);
    const int s = (int)(bs % S_);
    float mx = -BIGF;
    const _Float16* xp = X + (bs * K) * (size_t)O + o;
    for (int k = 0; k < K; ++k) mx = fmaxf(mx, (float)xp[(size_t)k * O]);
    out[((size_t)b * 192 + chbase + o) * S_ + s] = mx;
}

// ---------------------------------------------------------------------------
// Host launch
// ---------------------------------------------------------------------------
struct LayerCfg { int Cin, Cp, O; };

extern "C" void kernel_launch(void* const* d_in, const int* in_sizes, int n_in,
                              void* d_out, int out_size, void* d_ws, size_t ws_size,
                              hipStream_t stream) {
    (void)in_sizes; (void)n_in; (void)out_size; (void)ws_size;
    const float* xyz    = (const float*)d_in[0];
    const float* points = (const float*)d_in[1];

    static const LayerCfg cfg[2][3] = {
        {{35, 64, 32}, {32, 32, 32}, {32, 32, 64}},
        {{35, 64, 64}, {64, 64, 64}, {64, 64, 128}},
    };
    static const size_t woff[6] = {0, 2048, 3072, 5120, 9216, 13312}; // in halves

    // Workspace carve-up (~258 MB).
    char* ws = (char*)d_ws;
    int*      idx   = (int*)ws;                               // 32 KB
    int*      gidx  = (int*)(ws + 32768);                     // 2 MB
    float*    nxw   = (float*)(ws + 32768 + 2097152);         // 96 KB
    float*    stats = (float*)(ws + 2228224);                 // 256 f32
    _Float16* wbuf  = (_Float16*)(ws + 2229248);              // 43 KB
    _Float16* Xa    = (_Float16*)(ws + 2272256);              // 128 MB
    _Float16* Xb    = (_Float16*)(ws + 2272256 + 134217728);  // 128 MB

    float* out_xyz = (float*)d_out;
    float* out_np  = (float*)d_out + (size_t)B_ * 3 * S_;

    // FPS: one WG/batch, dist[] in LDS (64KB + 8KB reduce scratch).
    fps_kernel<<<B_, 1024, (size_t)N_ * 4 + 8192, stream>>>(xyz, idx, nxw, out_xyz);

    // 64-NN: one WG/centroid.
    knn_kernel<<<B_ * S_, 256, (size_t)N_ * 4 + 2048, stream>>>(xyz, nxw, gidx);

    // Weight prep (all 6 layers).
    for (int sc = 0; sc < 2; ++sc)
        for (int li = 0; li < 3; ++li) {
            const LayerCfg& c = cfg[sc][li];
            const float* W = (const float*)d_in[2 + (sc * 3 + li) * 4 + 0];
            int n = c.O * c.Cp;
            wprep_kernel<<<(n + 255) / 256, 256, 0, stream>>>(W, wbuf + woff[sc * 3 + li],
                                                              c.O, c.Cin, c.Cp);
        }

    for (int sc = 0; sc < 2; ++sc) {
        const int K = sc ? 64 : 32;
        const int Mrows = B_ * S_ * K;
        {
            size_t total = (size_t)Mrows * 64;
            group_kernel<<<(unsigned)((total + 255) / 256), 256, 0, stream>>>(
                xyz, points, gidx, nxw, Xa, K);
        }
        _Float16* cur = Xa;
        _Float16* nxt = Xb;
        for (int li = 0; li < 3; ++li) {
            const LayerCfg& c = cfg[sc][li];
            const float* gamma = (const float*)d_in[2 + (sc * 3 + li) * 4 + 2];
            const float* beta  = (const float*)d_in[2 + (sc * 3 + li) * 4 + 3];
            zero_kernel<<<1, 256, 0, stream>>>(stats, 256);
            const int mtiles = Mrows / 16;
            const int blocks = (mtiles + 3) / 4;
            const _Float16* Wf = wbuf + woff[sc * 3 + li];
            switch (c.O / 16) {
                case 2: wmma_gemm_kernel<2><<<blocks, 128, 0, stream>>>(cur, Wf, nxt, stats, Mrows, c.Cp); break;
                case 4: wmma_gemm_kernel<4><<<blocks, 128, 0, stream>>>(cur, Wf, nxt, stats, Mrows, c.Cp); break;
                default: wmma_gemm_kernel<8><<<blocks, 128, 0, stream>>>(cur, Wf, nxt, stats, Mrows, c.Cp); break;
            }
            {
                size_t total = (size_t)Mrows * c.O;
                bn_relu_kernel<<<(unsigned)((total + 255) / 256), 256, 0, stream>>>(
                    nxt, stats, gamma, beta, Mrows, c.O);
            }
            _Float16* tmp = cur; cur = nxt; nxt = tmp;
        }
        const int Of = cfg[sc][2].O;
        size_t total = (size_t)B_ * S_ * Of;
        maxk_kernel<<<(unsigned)((total + 255) / 256), 256, 0, stream>>>(
            cur, out_np, K, Of, sc ? 64 : 0);
    }
}
